// MoE_layer_megatron_wo_gate_46712064311449
// MI455X (gfx1250) — matrix operand order repaired and checked by reading, not compiled, not run on hardware
//
#include <hip/hip_runtime.h>
#include <hip/hip_bf16.h>
#include <cstdint>
#include <cstddef>

// MoE grouped MLP forward (Megatron GroupedMLP, no gate/bias), MI455X / gfx1250.
// fc1 grouped GEMM + exact gelu -> ws, fc2 grouped GEMM -> out.
// Precision: bf16x3 split (hi/lo) on v_wmma_f32_16x16x32_bf16, fp32 accumulate.
// Pipeline: register double-buffered global loads overlap HBM with WMMA.
// LDS layout: A rows natural [K0-31], B transposed [n][k]
// with paired-k 32-bit stores; every WMMA fragment is one contiguous 32B LDS
// load (2x ds_load_b128) straight into the WMMA operand registers.

typedef __attribute__((ext_vector_type(16))) __bf16 v16bf;
typedef __attribute__((ext_vector_type(8)))  float  v8f;

#define E_ 8
#define H_ 1024
#define F_ 4096
#define T_ 8192

__device__ __forceinline__ unsigned short f32_bf16_rne(float f) {
  unsigned int u = __builtin_bit_cast(unsigned int, f);
  u += 0x7FFFu + ((u >> 16) & 1u);          // round-to-nearest-even
  return (unsigned short)(u >> 16);
}

__device__ __forceinline__ float bf16_hi_f32(unsigned short h) {
  return __builtin_bit_cast(float, (unsigned int)h << 16);
}

__device__ __forceinline__ float gelu_exact(float x) {
  return 0.5f * x * (1.0f + erff(x * 0.70710678118654752f));
}

// A: [T,KDIM] fp32, rows sorted by expert.  B: [E,KDIM,NDIM] fp32.  C: [T,NDIM] fp32.
template<int KDIM, int NDIM, bool GELU>
__global__ __launch_bounds__(256) void grouped_gemm_bf16x3(
    const float* __restrict__ A,
    const float* __restrict__ B,
    const int*   __restrict__ tokens_per_expert,
    float* __restrict__ C)
{
  constexpr int TM = 128, TN = 128, TK = 32;
  constexpr int SA = TK + 8;            // LDS row stride (halves): 80B, kills bank conflicts
  constexpr int NT = NDIM / TN;

  __shared__ __align__(16) unsigned short AsH[TM * SA];
  __shared__ __align__(16) unsigned short AsL[TM * SA];
  __shared__ __align__(16) unsigned short BsH[TN * SA];   // transposed: [n][k]
  __shared__ __align__(16) unsigned short BsL[TN * SA];

  const int tid  = threadIdx.x;
  const int lane = tid & 31;
  const int wave = tid >> 5;
  const int wm   = wave >> 2;           // 0..1  (waves along M, 64 rows each)
  const int wn   = wave & 3;            // 0..3  (waves along N, 32 cols each)
  const int lr   = lane & 15;
  const int lh   = lane >> 4;           // lane half-group

  // staging coordinates (fixed per thread)
  const int ar  = tid >> 3;             // A row within 32-row slab (0..31)
  const int ac4 = (tid & 7) << 2;       // A col  (0..28)
  // A column: natural row layout [K0-31]
  const int apc4 = ac4;
  const int bk2 = (tid >> 5) << 1;      // B k-pair base within 16-row slab (0,2,..,14)
  const int bc4 = (tid & 31) << 2;      // B col  (0..124)

  int cnt[E_];
#pragma unroll
  for (int e = 0; e < E_; ++e) cnt[e] = tokens_per_expert[e];

  for (int tile = blockIdx.x; ; tile += gridDim.x) {
    // ragged flat-tile -> (expert, tile_m, tile_n); row tiles aligned to group start
    int rem = tile, rstart = 0, rows = 0, e = 0;
    bool found = false;
    for (e = 0; e < E_; ++e) {
      rows = cnt[e];
      int te = ((rows + TM - 1) / TM) * NT;
      if (rem < te) { found = true; break; }
      rem -= te; rstart += rows;
    }
    if (!found) return;                 // past total tile count -> all later ones are too

    const int row_end = rstart + rows;
    const int m0 = rstart + (rem / NT) * TM;
    const int n0 = (rem % NT) * TN;
    const float* __restrict__ Be = B + (size_t)e * KDIM * NDIM;

    v8f acc[4][2];
#pragma unroll
    for (int mi = 0; mi < 4; ++mi)
#pragma unroll
      for (int nj = 0; nj < 2; ++nj)
#pragma unroll
        for (int v = 0; v < 8; ++v) acc[mi][nj][v] = 0.0f;

    // register double-buffer for the next K-slab (8 x float4 per thread)
    float4 ga[4], gb0[2], gb1[2];
    auto load_tiles = [&](int k0) {
#pragma unroll
      for (int p = 0; p < 4; ++p) {
        const int grow = m0 + ar + p * 32;
        float4 va = {0.f, 0.f, 0.f, 0.f};
        if (grow < row_end)
          va = *(const float4*)(A + (size_t)grow * KDIM + k0 + ac4);
        ga[p] = va;
      }
#pragma unroll
      for (int p = 0; p < 2; ++p) {
        const float* bp = Be + (size_t)(k0 + bk2 + p * 16) * NDIM + n0 + bc4;
        gb0[p] = *(const float4*)(bp);
        gb1[p] = *(const float4*)(bp + NDIM);     // next k-row, same cols
      }
    };

    load_tiles(0);

    for (int k0 = 0; k0 < KDIM; k0 += TK) {
      __syncthreads();                  // previous compute done reading LDS
      // ---- convert staged registers -> hi/lo bf16 in LDS ----
#pragma unroll
      for (int p = 0; p < 4; ++p) {     // A tile: natural row layout, uint2 stores
        const float f[4] = {ga[p].x, ga[p].y, ga[p].z, ga[p].w};
        unsigned short h[4], l[4];
#pragma unroll
        for (int j = 0; j < 4; ++j) {
          h[j] = f32_bf16_rne(f[j]);
          l[j] = f32_bf16_rne(f[j] - bf16_hi_f32(h[j]));
        }
        uint2 ph, pl;
        ph.x = (unsigned)h[0] | ((unsigned)h[1] << 16);
        ph.y = (unsigned)h[2] | ((unsigned)h[3] << 16);
        pl.x = (unsigned)l[0] | ((unsigned)l[1] << 16);
        pl.y = (unsigned)l[2] | ((unsigned)l[3] << 16);
        const int r = ar + p * 32;
        *(uint2*)&AsH[r * SA + apc4] = ph;
        *(uint2*)&AsL[r * SA + apc4] = pl;
      }
#pragma unroll
      for (int p = 0; p < 2; ++p) {     // B tile: transposed [n][k], (k,k+1) pair stores
        const int k = bk2 + p * 16;
        const float f0[4] = {gb0[p].x, gb0[p].y, gb0[p].z, gb0[p].w};
        const float f1[4] = {gb1[p].x, gb1[p].y, gb1[p].z, gb1[p].w};
#pragma unroll
        for (int j = 0; j < 4; ++j) {
          const unsigned short h0 = f32_bf16_rne(f0[j]);
          const unsigned short h1 = f32_bf16_rne(f1[j]);
          const unsigned short l0 = f32_bf16_rne(f0[j] - bf16_hi_f32(h0));
          const unsigned short l1 = f32_bf16_rne(f1[j] - bf16_hi_f32(h1));
          *(unsigned int*)&BsH[(bc4 + j) * SA + k] = (unsigned)h0 | ((unsigned)h1 << 16);
          *(unsigned int*)&BsL[(bc4 + j) * SA + k] = (unsigned)l0 | ((unsigned)l1 << 16);
        }
      }
      __syncthreads();

      // issue next K-slab global loads; s_wait_loadcnt lands in next iter's
      // convert phase, so HBM latency hides behind the WMMA burst below
      if (k0 + TK < KDIM) load_tiles(k0 + TK);

      // ---- B fragments (32x16 bf16): lane = col n (lr), K = lh*16 + 0..15 ----
      //      contiguous 32B -> direct v16bf load (2x ds_load_b128)
      v16bf bh[2], bl[2];
#pragma unroll
      for (int nj = 0; nj < 2; ++nj) {
        const int n = wn * 32 + nj * 16 + lr;
        bh[nj] = *(const v16bf*)&BsH[n * SA + lh * 16];
        bl[nj] = *(const v16bf*)&BsL[n * SA + lh * 16];
      }

      // ---- A fragments (16x32 bf16, natural rows => contiguous 32B) + 24 WMMAs ----
#pragma unroll
      for (int mi = 0; mi < 4; ++mi) {
        const int r = wm * 64 + mi * 16 + lr;       // row in tile
        v16bf ah = *(const v16bf*)&AsH[r * SA + lh * 16];
        v16bf al = *(const v16bf*)&AsL[r * SA + lh * 16];
#pragma unroll
        for (int nj = 0; nj < 2; ++nj) {
          acc[mi][nj] = __builtin_amdgcn_wmma_f32_16x16x32_bf16(
              false, ah, false, bh[nj], (short)0, acc[mi][nj], false, false);
          acc[mi][nj] = __builtin_amdgcn_wmma_f32_16x16x32_bf16(
              false, ah, false, bl[nj], (short)0, acc[mi][nj], false, false);
          acc[mi][nj] = __builtin_amdgcn_wmma_f32_16x16x32_bf16(
              false, al, false, bh[nj], (short)0, acc[mi][nj], false, false);
        }
      }
    }

    // ---- epilogue: C/D layout -> lanes 0-15 N, VGPR v -> M = lh*8 + v ----
#pragma unroll
    for (int mi = 0; mi < 4; ++mi) {
#pragma unroll
      for (int nj = 0; nj < 2; ++nj) {
        const int col  = n0 + wn * 32 + nj * 16 + lr;
        const int rowb = m0 + wm * 64 + mi * 16 + lh * 8;
#pragma unroll
        for (int v = 0; v < 8; ++v) {
          const int row = rowb + v;
          if (row < row_end) {
            float xv = acc[mi][nj][v];
            if (GELU) xv = gelu_exact(xv);
            C[(size_t)row * NDIM + col] = xv;
          }
        }
      }
    }
  }
}

extern "C" void kernel_launch(void* const* d_in, const int* in_sizes, int n_in,
                              void* d_out, int out_size, void* d_ws, size_t ws_size,
                              hipStream_t stream) {
  const float* x   = (const float*)d_in[0];   // [T,H]
  const int*   tpe = (const int*)d_in[1];     // [E]
  const float* w1  = (const float*)d_in[2];   // [E,H,F]
  const float* w2  = (const float*)d_in[3];   // [E,F,H]
  float* out = (float*)d_out;                 // [T,H]
  float* act = (float*)d_ws;                  // [T,F] fp32 intermediate (128 MB of ws)

  // Upper bounds on ragged tile counts (device-side group sizes): extra blocks
  // scan, find no tile, and exit immediately.
  const int grid1 = (T_ / 128 + E_) * (F_ / 128);   // 2304
  const int grid2 = (T_ / 128 + E_) * (H_ / 128);   // 576

  grouped_gemm_bf16x3<H_, F_, true ><<<grid1, 256, 0, stream>>>(x,   w1, tpe, act);
  grouped_gemm_bf16x3<F_, H_, false><<<grid2, 256, 0, stream>>>(act, w2, tpe, out);
}
